// SVDLlamaBlock_85529978733252
// MI455X (gfx1250) — compile-verified, matrix operand order checked by reading, not run on hardware
//
#include <hip/hip_runtime.h>
#include <stdint.h>

// ---- problem constants (match reference) ----
#define T_SEQ   2048
#define D_MODEL 2048
#define N_H     32
#define N_HK    8
#define D_HEAD  64
#define R_Q     32
#define R_KV    32
#define R_O     512
#define R_FF    512
#define D_INTER 8192

typedef __bf16 bf16_t;
typedef __attribute__((ext_vector_type(16))) __bf16        v16bf;
typedef __attribute__((ext_vector_type(8)))  float         v8f;
typedef __attribute__((ext_vector_type(4)))  int           v4i;
typedef __attribute__((ext_vector_type(8)))  int           v8i;
typedef __attribute__((ext_vector_type(4)))  unsigned int  v4u;

#if defined(__has_builtin)
#if __has_builtin(__builtin_amdgcn_tensor_load_to_lds) && __has_builtin(__builtin_amdgcn_s_wait_tensorcnt)
#define USE_TDM 1
#endif
#endif
#ifndef USE_TDM
#define USE_TDM 0
#endif

__device__ __forceinline__ bf16_t f2bf(float f) {
  unsigned u = __float_as_uint(f);
  unsigned r = (u + 0x7FFFu + ((u >> 16) & 1u)) >> 16;   // round-to-nearest-even
  unsigned short s = (unsigned short)r;
  return __builtin_bit_cast(bf16_t, s);
}
__device__ __forceinline__ float bf2f(bf16_t b) {
  unsigned short s = __builtin_bit_cast(unsigned short, b);
  return __uint_as_float(((unsigned)s) << 16);
}
__device__ __forceinline__ v8f vzero() {
  v8f z = {0.f,0.f,0.f,0.f,0.f,0.f,0.f,0.f};
  return z;
}

union FragU { v16bf bf; v4i q[2]; };

// 16x32 bf16 fragment from a row-major [rows, ld] matrix (A-layout; B loads the
// transpose-symmetric layout identically since we compute A * B^T).
__device__ __forceinline__ v16bf load_frag(const bf16_t* base, int ld) {
  int lane = threadIdx.x & 31;
  const bf16_t* p = base + (long)(lane & 15) * ld + ((lane >> 4) << 3);
  FragU u;
  u.q[0] = *(const v4i*)(p);
  u.q[1] = *(const v4i*)(p + 16);
  return u.bf;
}

__device__ __forceinline__ v8f wmma_bf16(v16bf a, v16bf b, v8f c) {
  return __builtin_amdgcn_wmma_f32_16x16x32_bf16(false, a, false, b, (short)0, c, false, false);
}

// ---------------- elementwise kernels ----------------
__global__ __launch_bounds__(256) void cast_kernel(const float* __restrict__ in,
                                                   bf16_t* __restrict__ out, long n) {
  long i = (long)blockIdx.x * 256 + threadIdx.x;
  if (i < n) out[i] = f2bf(in[i]);
}

__global__ __launch_bounds__(256) void silu_mul_kernel(const bf16_t* __restrict__ g,
                                                       const bf16_t* __restrict__ u,
                                                       bf16_t* __restrict__ o, long n) {
  long i = (long)blockIdx.x * 256 + threadIdx.x;
  if (i < n) {
    float gv = bf2f(g[i]), uv = bf2f(u[i]);
    float s = gv / (1.f + __expf(-gv));
    o[i] = f2bf(s * uv);
  }
}

__global__ __launch_bounds__(256) void rmsnorm_kernel(const float* __restrict__ x,
                                                      const float* __restrict__ w,
                                                      bf16_t* __restrict__ out) {
  const int t = blockIdx.x;
  const float* row = x + (long)t * D_MODEL;
  float ss = 0.f;
  for (int i = threadIdx.x; i < D_MODEL; i += 256) { float v = row[i]; ss += v * v; }
  #pragma unroll
  for (int o = 16; o > 0; o >>= 1) ss += __shfl_xor(ss, o, 32);
  __shared__ float red[8];
  __shared__ float inv_s;
  if ((threadIdx.x & 31) == 0) red[threadIdx.x >> 5] = ss;
  __syncthreads();
  if (threadIdx.x == 0) {
    float s = 0.f;
    #pragma unroll
    for (int i = 0; i < 8; ++i) s += red[i];
    inv_s = rsqrtf(s / (float)D_MODEL + 1e-5f);
  }
  __syncthreads();
  float inv = inv_s;
  bf16_t* orow = out + (long)t * D_MODEL;
  for (int i = threadIdx.x; i < D_MODEL; i += 256)
    orow[i] = f2bf(row[i] * inv * w[i]);
}

// RoPE: in f32 [Hn][T][64] -> out bf16
__global__ __launch_bounds__(256) void rope_kernel(const float* __restrict__ in,
                                                   bf16_t* __restrict__ out, int Hn) {
  long idx = (long)blockIdx.x * 256 + threadIdx.x;
  long total = (long)Hn * T_SEQ * 32;
  if (idx >= total) return;
  int  i  = (int)(idx & 31);
  long ht = idx >> 5;
  int  t  = (int)(ht % T_SEQ);
  long h  = ht / T_SEQ;
  float inv_freq = __powf(10000.0f, -(float)i / 32.0f);
  float sn, cs; __sincosf((float)t * inv_freq, &sn, &cs);
  const float* p = in + (h * T_SEQ + t) * D_HEAD;
  float x1 = p[i], x2 = p[i + 32];
  bf16_t* o = out + (h * T_SEQ + t) * D_HEAD;
  o[i]      = f2bf(x1 * cs - x2 * sn);
  o[i + 32] = f2bf(x2 * cs + x1 * sn);
}

// ------------- WMMA GEMM: C = A(MxK) * B(NxK)^T [+ Cadd], optional batch -------------
// Wave tile = 32 x (BN*16); 2x2 waves per block -> block tile 64 x (BN*32).
// Software-pipelined: next K-step fragments loaded before current WMMAs issue.
template<int BN>
__global__ __launch_bounds__(128) void gemm_bf16_kernel(
    const bf16_t* __restrict__ A, const bf16_t* __restrict__ B, void* __restrict__ Cv,
    const float* __restrict__ Cadd, int M, int N, int K,
    int lda, int ldb, int ldc, long Abs, long Bbs, long Cbs, int out_bf16)
{
  const int z = blockIdx.z;
  const bf16_t* Az = A + (long)z * Abs;
  const bf16_t* Bz = B + (long)z * Bbs;
  const int wave = threadIdx.x >> 5;
  const int lane = threadIdx.x & 31;
  const int m0 = blockIdx.y * 64 + (wave >> 1) * 32;
  const int n0 = blockIdx.x * (BN * 32) + (wave & 1) * (BN * 16);
  if (m0 >= M || n0 >= N) return;

  const bf16_t* Ap[2];
  Ap[0] = Az + (long)m0 * lda;
  Ap[1] = Az + (long)(m0 + 16) * lda;
  const bf16_t* Bp[BN];
  #pragma unroll
  for (int j = 0; j < BN; ++j) Bp[j] = Bz + (long)(n0 + 16 * j) * ldb;

  v8f acc[2][BN];
  #pragma unroll
  for (int i = 0; i < 2; ++i)
    #pragma unroll
    for (int j = 0; j < BN; ++j) acc[i][j] = vzero();

  v16bf a_cur[2], b_cur[BN];
  #pragma unroll
  for (int i = 0; i < 2; ++i) a_cur[i] = load_frag(Ap[i], lda);
  #pragma unroll
  for (int j = 0; j < BN; ++j) b_cur[j] = load_frag(Bp[j], ldb);

  for (int k = 32; k <= K; k += 32) {
    v16bf a_nxt[2], b_nxt[BN];
    const bool more = (k < K);
    if (more) {
      #pragma unroll
      for (int i = 0; i < 2; ++i) a_nxt[i] = load_frag(Ap[i] + k, lda);
      #pragma unroll
      for (int j = 0; j < BN; ++j) b_nxt[j] = load_frag(Bp[j] + k, ldb);
      if (k + 32 < K) {                       // -> global_prefetch_b8
        __builtin_prefetch(Ap[0] + k + 32, 0, 1);
        __builtin_prefetch(Bp[0] + k + 32, 0, 1);
      }
    }
    #pragma unroll
    for (int i = 0; i < 2; ++i)
      #pragma unroll
      for (int j = 0; j < BN; ++j)
        acc[i][j] = wmma_bf16(a_cur[i], b_cur[j], acc[i][j]);
    if (more) {
      #pragma unroll
      for (int i = 0; i < 2; ++i) a_cur[i] = a_nxt[i];
      #pragma unroll
      for (int j = 0; j < BN; ++j) b_cur[j] = b_nxt[j];
    }
  }

  const int col = lane & 15, rb = (lane >> 4) * 8;
  #pragma unroll
  for (int mi = 0; mi < 2; ++mi)
    #pragma unroll
    for (int ni = 0; ni < BN; ++ni)
      #pragma unroll
      for (int e = 0; e < 8; ++e) {
        int m = m0 + mi * 16 + rb + e;
        int n = n0 + ni * 16 + col;
        float v = acc[mi][ni][e];
        if (Cadd) v += Cadd[(long)m * ldc + n];
        long ci = (long)z * Cbs + (long)m * ldc + n;
        if (out_bf16) ((bf16_t*)Cv)[ci] = f2bf(v);
        else          ((float*)Cv)[ci]  = v;
      }
}

// ---------------- Flash attention (causal, GQA rep=4), WMMA QK^T and PV ----------------
// grid (T/64, H), block 128. K tile staged into LDS by the Tensor Data Mover
// (tensor_load_to_lds + s_wait_tensorcnt); V tile transposed into LDS.
__global__ __launch_bounds__(128) void attn_kernel(
    const bf16_t* __restrict__ Q, const bf16_t* __restrict__ K,
    const bf16_t* __restrict__ V, bf16_t* __restrict__ O)
{
  __shared__ __align__(16) bf16_t Vt[64][32];     // V^T tile: [d][key]
  __shared__ __align__(16) bf16_t Pl[4][16][32];  // per-wave P tile (A-layout staging)
#if USE_TDM
  __shared__ __align__(16) bf16_t Kt[32][64];     // K tile, TDM-loaded
#endif

  const int q0   = blockIdx.x * 64;
  const int h    = blockIdx.y;
  const int hk   = h >> 2;                        // REP = H/Hk = 4
  const int wq   = threadIdx.x >> 5;
  const int lane = threadIdx.x & 31;
  const int col  = lane & 15;
  const int rb   = (lane >> 4) * 8;
  const int qw   = q0 + wq * 16;
  const float scale = 0.125f;                     // 1/sqrt(64)
  const float NEG = -__builtin_inff();

  const bf16_t* Qh = Q + (long)h  * T_SEQ * D_HEAD;
  const bf16_t* Kh = K + (long)hk * T_SEQ * D_HEAD;
  const bf16_t* Vh = V + (long)hk * T_SEQ * D_HEAD;

  v16bf aq0 = load_frag(Qh + (long)qw * D_HEAD + 0,  D_HEAD);
  v16bf aq1 = load_frag(Qh + (long)qw * D_HEAD + 32, D_HEAD);

  float mrow[8], lrow[8];
  v8f oacc[4];
  #pragma unroll
  for (int e = 0; e < 8; ++e) { mrow[e] = NEG; lrow[e] = 0.f; }
  #pragma unroll
  for (int nt = 0; nt < 4; ++nt) oacc[nt] = vzero();

  const int kend = q0 + 63;
  for (int kb = 0; kb <= kend; kb += 32) {
#if USE_TDM
    // TDM: DMA K tile [32 keys][64] (bf16) into LDS; wave 0 issues, all waves sync.
    if (threadIdx.x < 32) {
      unsigned lds_off = (unsigned)(size_t)(&Kt[0][0]);
      unsigned long long ga = (unsigned long long)(size_t)(Kh + (long)kb * D_HEAD);
      v4u g0;
      g0[0] = 1u;                                        // count = 1 (valid D#)
      g0[1] = lds_off;                                   // lds_addr (bytes)
      g0[2] = (unsigned)ga;                              // global_addr[31:0]
      g0[3] = (unsigned)((ga >> 32) & 0x1FFFFFFu) | (2u << 30); // addr[56:32] | type=2
      v8i g1;
      g1[0] = (int)(1u << 16);                           // data_size=1 (2B), mask=0
      g1[1] = (int)((unsigned)D_HEAD << 16);             // tensor_dim0 = 64 (low16)
      g1[2] = (int)((unsigned)T_SEQ << 16);              // tensor_dim1 = 2048 (low16)
      g1[3] = (int)((unsigned)D_HEAD << 16);             // tile_dim0 = 64
      g1[4] = 32;                                        // tile_dim1 = 32, tile_dim2 = 0
      g1[5] = D_HEAD;                                    // tensor_dim0_stride = 64
      g1[6] = 0;
      g1[7] = 0;
      v4i gz = {0, 0, 0, 0};
      v8i gz8 = {0, 0, 0, 0, 0, 0, 0, 0};
      // amdgpu-toolchain (clang-23) 6-arg form: (g0, g1, g2, g3, g4, cpol)
      __builtin_amdgcn_tensor_load_to_lds(g0, g1, gz, gz, gz8, 0);
      __builtin_amdgcn_s_wait_tensorcnt(0);
    }
#endif
    // cooperative transposed load of V tile [32 keys][64] -> Vt[64][32]
    for (int idx = threadIdx.x; idx < 32 * 64; idx += 128) {
      int kk = idx >> 6, d = idx & 63;
      Vt[d][kk] = Vh[(long)(kb + kk) * D_HEAD + d];
    }
    __syncthreads();

    const bool active = (kb <= qw + 15);          // wave-uniform
    if (active) {
#if USE_TDM
      const bf16_t* Kp = &Kt[0][0];               // LDS-resident K tile
#else
      const bf16_t* Kp = Kh + (long)kb * D_HEAD;  // direct from global
#endif
      // S = Q K^T  (16x32 scores, two 16x16 accumulators)
      v8f s0 = vzero(), s1 = vzero();
      v16bf b00 = load_frag(Kp + 0 * D_HEAD + 0,  D_HEAD);
      v16bf b01 = load_frag(Kp + 0 * D_HEAD + 32, D_HEAD);
      v16bf b10 = load_frag(Kp + 16 * D_HEAD + 0,  D_HEAD);
      v16bf b11 = load_frag(Kp + 16 * D_HEAD + 32, D_HEAD);
      s0 = wmma_bf16(aq0, b00, s0);
      s0 = wmma_bf16(aq1, b01, s0);
      s1 = wmma_bf16(aq0, b10, s1);
      s1 = wmma_bf16(aq1, b11, s1);

      // online softmax, row = qw + rb + e held across each 16-lane half
      #pragma unroll
      for (int e = 0; e < 8; ++e) {
        int qr = qw + rb + e;
        float a = s0[e] * scale; if (kb + col      > qr) a = NEG;
        float b = s1[e] * scale; if (kb + 16 + col > qr) b = NEG;
        float mx = fmaxf(a, b);
        #pragma unroll
        for (int o = 8; o > 0; o >>= 1) mx = fmaxf(mx, __shfl_xor(mx, o, 32));
        float mold = mrow[e];
        float mnew = fmaxf(mold, mx);
        float so, pa, pb;
        if (mnew == NEG) { so = 1.f; pa = 0.f; pb = 0.f; }
        else { so = __expf(mold - mnew); pa = __expf(a - mnew); pb = __expf(b - mnew); }
        float ps = pa + pb;
        #pragma unroll
        for (int o = 8; o > 0; o >>= 1) ps += __shfl_xor(ps, o, 32);
        lrow[e] = lrow[e] * so + ps;
        mrow[e] = mnew;
        #pragma unroll
        for (int nt = 0; nt < 4; ++nt) oacc[nt][e] = oacc[nt][e] * so;
        Pl[wq][rb + e][col]      = f2bf(pa);
        Pl[wq][rb + e][col + 16] = f2bf(pb);
      }

      // O += P * V  (LDS A-frag for P, LDS B-frags for V^T)
      v16bf pf = load_frag(&Pl[wq][0][0], 32);
      #pragma unroll
      for (int nt = 0; nt < 4; ++nt) {
        v16bf vf = load_frag(&Vt[nt * 16][0], 32);
        oacc[nt] = wmma_bf16(pf, vf, oacc[nt]);
      }
    }
    __syncthreads();
  }

  #pragma unroll
  for (int e = 0; e < 8; ++e) {
    float inv = 1.f / lrow[e];
    int t = qw + rb + e;
    #pragma unroll
    for (int nt = 0; nt < 4; ++nt)
      O[(long)t * D_MODEL + h * D_HEAD + nt * 16 + col] = f2bf(oacc[nt][e] * inv);
  }
}

// ---------------- host side ----------------
static inline void cast_w(hipStream_t s, const float* in, bf16_t* out, long n) {
  cast_kernel<<<(unsigned)((n + 255) / 256), 256, 0, s>>>(in, out, n);
}
// wide=1 -> 64x128 block tile (N % 128 == 0); wide=0 -> 64x64 block tile
static inline void gemm(hipStream_t s, const bf16_t* A, const bf16_t* B, void* C,
                        const float* Cadd, int M, int N, int K,
                        int lda, int ldb, int ldc,
                        long Abs, long Bbs, long Cbs, int batch, int out_bf16, int wide) {
  if (wide) {
    dim3 g(N / 128, M / 64, batch), b(128);
    gemm_bf16_kernel<4><<<g, b, 0, s>>>(A, B, C, Cadd, M, N, K, lda, ldb, ldc, Abs, Bbs, Cbs, out_bf16);
  } else {
    dim3 g(N / 64, M / 64, batch), b(128);
    gemm_bf16_kernel<2><<<g, b, 0, s>>>(A, B, C, Cadd, M, N, K, lda, ldb, ldc, Abs, Bbs, Cbs, out_bf16);
  }
}

extern "C" void kernel_launch(void* const* d_in, const int* in_sizes, int n_in,
                              void* d_out, int out_size, void* d_ws, size_t ws_size,
                              hipStream_t stream) {
  (void)in_sizes; (void)n_in; (void)out_size; (void)ws_size;
  const float* hidden = (const float*)d_in[0];
  const float* ln1    = (const float*)d_in[1];
  const float* ln2    = (const float*)d_in[2];
  const float* q_Us   = (const float*)d_in[3];
  const float* q_V    = (const float*)d_in[4];
  const float* k_Us   = (const float*)d_in[5];
  const float* k_V    = (const float*)d_in[6];
  const float* v_Us   = (const float*)d_in[7];
  const float* v_V    = (const float*)d_in[8];
  const float* o_Us   = (const float*)d_in[9];
  const float* o_V    = (const float*)d_in[10];
  const float* g_Us   = (const float*)d_in[11];
  const float* g_V    = (const float*)d_in[12];
  const float* u_Us   = (const float*)d_in[13];
  const float* u_V    = (const float*)d_in[14];
  const float* d_Usp  = (const float*)d_in[15];
  const float* d_Vp   = (const float*)d_in[16];

  char* base = (char*)d_ws;
  size_t off = 0;
  auto alloc = [&](long elems, size_t esz) -> void* {
    off = (off + 255) & ~(size_t)255;
    void* p = base + off;
    off += (size_t)elems * esz;
    return p;
  };

  const long szD = (long)T_SEQ * D_MODEL;
  // bf16 weight copies (L2-resident, ~42 MB total)
  bf16_t* qV_b  = (bf16_t*)alloc((long)N_H  * R_Q  * D_MODEL, 2);
  bf16_t* kV_b  = (bf16_t*)alloc((long)N_HK * R_KV * D_MODEL, 2);
  bf16_t* vV_b  = (bf16_t*)alloc((long)N_HK * R_KV * D_MODEL, 2);
  bf16_t* qUs_b = (bf16_t*)alloc((long)N_H  * D_HEAD * R_Q, 2);
  bf16_t* kUs_b = (bf16_t*)alloc((long)N_HK * D_HEAD * R_KV, 2);
  bf16_t* vUs_b = (bf16_t*)alloc((long)N_HK * D_HEAD * R_KV, 2);
  bf16_t* oV_b  = (bf16_t*)alloc((long)R_O * D_MODEL, 2);
  bf16_t* oUs_b = (bf16_t*)alloc((long)D_MODEL * R_O, 2);
  bf16_t* gV_b  = (bf16_t*)alloc((long)R_FF * D_MODEL, 2);
  bf16_t* uV_b  = (bf16_t*)alloc((long)R_FF * D_MODEL, 2);
  bf16_t* gUs_b = (bf16_t*)alloc((long)D_INTER * R_FF, 2);
  bf16_t* uUs_b = (bf16_t*)alloc((long)D_INTER * R_FF, 2);
  bf16_t* dV_b  = (bf16_t*)alloc((long)R_FF * D_INTER, 2);
  bf16_t* dUs_b = (bf16_t*)alloc((long)D_MODEL * R_FF, 2);
  // activations
  bf16_t* x_b    = (bf16_t*)alloc(szD, 2);
  bf16_t* Pq_b   = (bf16_t*)alloc((long)T_SEQ * N_H  * R_Q, 2);
  bf16_t* Pk_b   = (bf16_t*)alloc((long)T_SEQ * N_HK * R_KV, 2);
  bf16_t* Pv_b   = (bf16_t*)alloc((long)T_SEQ * N_HK * R_KV, 2);
  float*  qf     = (float*) alloc((long)N_H  * T_SEQ * D_HEAD, 4);
  float*  kf     = (float*) alloc((long)N_HK * T_SEQ * D_HEAD, 4);
  bf16_t* qbuf   = (bf16_t*)alloc((long)N_H  * T_SEQ * D_HEAD, 2);
  bf16_t* kbuf   = (bf16_t*)alloc((long)N_HK * T_SEQ * D_HEAD, 2);
  bf16_t* vbuf   = (bf16_t*)alloc((long)N_HK * T_SEQ * D_HEAD, 2);
  bf16_t* attn_b = (bf16_t*)alloc(szD, 2);
  bf16_t* Pao_b  = (bf16_t*)alloc((long)T_SEQ * R_O, 2);
  float*  hbuf   = (float*) alloc(szD, 4);
  bf16_t* x2_b   = (bf16_t*)alloc(szD, 2);
  bf16_t* Pg_b   = (bf16_t*)alloc((long)T_SEQ * R_FF, 2);
  bf16_t* Pu_b   = (bf16_t*)alloc((long)T_SEQ * R_FF, 2);
  bf16_t* gate_b = (bf16_t*)alloc((long)T_SEQ * D_INTER, 2);
  bf16_t* up_b   = (bf16_t*)alloc((long)T_SEQ * D_INTER, 2);
  bf16_t* act_b  = (bf16_t*)alloc((long)T_SEQ * D_INTER, 2);
  bf16_t* Pd_b   = (bf16_t*)alloc((long)T_SEQ * R_FF, 2);

  // weight casts
  cast_w(stream, q_V,  qV_b,  (long)N_H  * R_Q  * D_MODEL);
  cast_w(stream, k_V,  kV_b,  (long)N_HK * R_KV * D_MODEL);
  cast_w(stream, v_V,  vV_b,  (long)N_HK * R_KV * D_MODEL);
  cast_w(stream, q_Us, qUs_b, (long)N_H  * D_HEAD * R_Q);
  cast_w(stream, k_Us, kUs_b, (long)N_HK * D_HEAD * R_KV);
  cast_w(stream, v_Us, vUs_b, (long)N_HK * D_HEAD * R_KV);
  cast_w(stream, o_V,  oV_b,  (long)R_O * D_MODEL);
  cast_w(stream, o_Us, oUs_b, (long)D_MODEL * R_O);
  cast_w(stream, g_V,  gV_b,  (long)R_FF * D_MODEL);
  cast_w(stream, u_V,  uV_b,  (long)R_FF * D_MODEL);
  cast_w(stream, g_Us, gUs_b, (long)D_INTER * R_FF);
  cast_w(stream, u_Us, uUs_b, (long)D_INTER * R_FF);
  cast_w(stream, d_Vp, dV_b,  (long)R_FF * D_INTER);
  cast_w(stream, d_Usp, dUs_b,(long)D_MODEL * R_FF);

  // ---- block ----
  rmsnorm_kernel<<<T_SEQ, 256, 0, stream>>>(hidden, ln1, x_b);

  // low-rank projections: P = x @ V^T
  gemm(stream, x_b, qV_b, Pq_b, nullptr, T_SEQ, N_H * R_Q,  D_MODEL, D_MODEL, D_MODEL, N_H * R_Q,  0, 0, 0, 1, 1, 1);
  gemm(stream, x_b, kV_b, Pk_b, nullptr, T_SEQ, N_HK * R_KV, D_MODEL, D_MODEL, D_MODEL, N_HK * R_KV, 0, 0, 0, 1, 1, 1);
  gemm(stream, x_b, vV_b, Pv_b, nullptr, T_SEQ, N_HK * R_KV, D_MODEL, D_MODEL, D_MODEL, N_HK * R_KV, 0, 0, 0, 1, 1, 1);

  // per-head up-projections: q/k/v = P @ Us^T  (batched over heads, N=64 -> narrow tile)
  gemm(stream, Pq_b, qUs_b, qf,   nullptr, T_SEQ, D_HEAD, R_Q,  N_H * R_Q,  R_Q,  D_HEAD,
       R_Q,  (long)D_HEAD * R_Q,  (long)T_SEQ * D_HEAD, N_H, 0, 0);
  gemm(stream, Pk_b, kUs_b, kf,   nullptr, T_SEQ, D_HEAD, R_KV, N_HK * R_KV, R_KV, D_HEAD,
       R_KV, (long)D_HEAD * R_KV, (long)T_SEQ * D_HEAD, N_HK, 0, 0);
  gemm(stream, Pv_b, vUs_b, vbuf, nullptr, T_SEQ, D_HEAD, R_KV, N_HK * R_KV, R_KV, D_HEAD,
       R_KV, (long)D_HEAD * R_KV, (long)T_SEQ * D_HEAD, N_HK, 1, 0);

  // RoPE (f32 -> bf16)
  rope_kernel<<<(unsigned)(((long)N_H  * T_SEQ * 32 + 255) / 256), 256, 0, stream>>>(qf, qbuf, N_H);
  rope_kernel<<<(unsigned)(((long)N_HK * T_SEQ * 32 + 255) / 256), 256, 0, stream>>>(kf, kbuf, N_HK);

  // causal flash attention, GQA rep=4
  attn_kernel<<<dim3(T_SEQ / 64, N_H), 128, 0, stream>>>(qbuf, kbuf, vbuf, attn_b);

  // output projection + residual: h = hidden + (attn @ oV^T) @ oUs^T
  gemm(stream, attn_b, oV_b, Pao_b, nullptr, T_SEQ, R_O, D_MODEL, D_MODEL, D_MODEL, R_O, 0, 0, 0, 1, 1, 1);
  gemm(stream, Pao_b, oUs_b, hbuf, hidden, T_SEQ, D_MODEL, R_O, R_O, R_O, D_MODEL, 0, 0, 0, 1, 0, 1);

  // FFN
  rmsnorm_kernel<<<T_SEQ, 256, 0, stream>>>(hbuf, ln2, x2_b);
  gemm(stream, x2_b, gV_b, Pg_b, nullptr, T_SEQ, R_FF, D_MODEL, D_MODEL, D_MODEL, R_FF, 0, 0, 0, 1, 1, 1);
  gemm(stream, x2_b, uV_b, Pu_b, nullptr, T_SEQ, R_FF, D_MODEL, D_MODEL, D_MODEL, R_FF, 0, 0, 0, 1, 1, 1);
  gemm(stream, Pg_b, gUs_b, gate_b, nullptr, T_SEQ, D_INTER, R_FF, R_FF, R_FF, D_INTER, 0, 0, 0, 1, 1, 1);
  gemm(stream, Pu_b, uUs_b, up_b,   nullptr, T_SEQ, D_INTER, R_FF, R_FF, R_FF, D_INTER, 0, 0, 0, 1, 1, 1);
  silu_mul_kernel<<<(unsigned)(((long)T_SEQ * D_INTER + 255) / 256), 256, 0, stream>>>(
      gate_b, up_b, act_b, (long)T_SEQ * D_INTER);
  gemm(stream, act_b, dV_b, Pd_b, nullptr, T_SEQ, R_FF, D_INTER, D_INTER, D_INTER, R_FF, 0, 0, 0, 1, 1, 1);
  // out = h + Pd @ dUs^T  (f32, straight into d_out)
  gemm(stream, Pd_b, dUs_b, (float*)d_out, hbuf, T_SEQ, D_MODEL, R_FF, R_FF, R_FF, D_MODEL, 0, 0, 0, 1, 0, 1);
}
